// HeteroGAT_47261820125232
// MI455X (gfx1250) — matrix operand order repaired and checked by reading, not compiled, not run on hardware
//
#include <hip/hip_runtime.h>
#include <stdint.h>
#include <stddef.h>

#define HEADS 8
#define NEG_SLOPE 0.2f
#define LN_EPS 1e-5f

typedef __attribute__((ext_vector_type(16))) __bf16 v16bf;
typedef __attribute__((ext_vector_type(8)))  __bf16 v8bf;
typedef __attribute__((ext_vector_type(8)))  float  v8f;

static __device__ __forceinline__ unsigned short f32_to_bf16_bits(float f) {
  unsigned u = __builtin_bit_cast(unsigned, f);
  u += 0x7FFFu + ((u >> 16) & 1u);            // round-to-nearest-even
  return (unsigned short)(u >> 16);
}
// order-preserving f32 <-> u32 mapping for atomicMax-based segment max
static __device__ __forceinline__ unsigned enc_ordered(float f) {
  unsigned u = __builtin_bit_cast(unsigned, f);
  return (u & 0x80000000u) ? ~u : (u | 0x80000000u);
}
static __device__ __forceinline__ float dec_ordered(unsigned u) {
  unsigned v = (u & 0x80000000u) ? (u & 0x7FFFFFFFu) : ~u;
  return __builtin_bit_cast(float, v);
}

// ------------------------------ kernels ------------------------------

__global__ void k_zero(float* __restrict__ p, int count) {
  int i = blockIdx.x * blockDim.x + threadIdx.x;
  int st = gridDim.x * blockDim.x;
  for (; i < count; i += st) p[i] = 0.0f;
}

__global__ void k_f32_to_bf16(const float* __restrict__ src,
                              unsigned short* __restrict__ dst, int count) {
  int i = blockIdx.x * blockDim.x + threadIdx.x;
  int st = gridDim.x * blockDim.x;
  for (; i < count; i += st) dst[i] = f32_to_bf16_bits(src[i]);
}

// Pack W[K,64] (f32, row-major) into WMMA B-fragment layout so each lane can
// load its 16-bf16 fragment contiguously:
//   out[((ks*4+ct)*32 + lane)*16 + j] =
//       bf16( W[(ks*32 + 8*(lane>>4) + (j&7) + 16*(j>>3))*64 + ct*16 + (lane&15)] )
__global__ void k_pack_b(const float* __restrict__ W,
                         unsigned short* __restrict__ out, int K) {
  int t = blockIdx.x * blockDim.x + threadIdx.x;
  const int total = K * 64;
  if (t >= total) return;
  const int j    = t & 15;
  const int lane = (t >> 4) & 31;
  const int ct   = (t >> 9) & 3;
  const int ks   = t >> 11;
  const int krow = ks * 32 + 8 * (lane >> 4) + (j & 7) + 16 * (j >> 3);
  const int col  = ct * 16 + (lane & 15);
  out[t] = f32_to_bf16_bits(W[(size_t)krow * 64 + col]);
}

// h[n,64] = xbf[n,K] @ W + b  via v_wmma_f32_16x16x32_bf16 (W pre-packed)
__global__ void k_proj_wmma(const unsigned short* __restrict__ xbf,
                            const unsigned short* __restrict__ wpack,
                            const float* __restrict__ bias,
                            float* __restrict__ hout,
                            int n, int K) {
  const int lane = threadIdx.x & 31;
  const int wave = (blockIdx.x * blockDim.x + threadIdx.x) >> 5;
  const int row0 = wave * 16;
  if (row0 >= n) return;                      // wave-uniform: EXEC stays all-1 for WMMA
  const int half = lane >> 4;
  const int lr   = lane & 15;

  v8f acc[4] = {};
  const unsigned short* arow = xbf + (size_t)(row0 + lr) * K;
  for (int k0 = 0; k0 < K; k0 += 32) {
    // A (16x32 bf16): lane holds K = k0+8*half+{0..7} and k0+16+8*half+{0..7}
    v8bf alo = *(const v8bf*)(arow + k0 + 8 * half);
    v8bf ahi = *(const v8bf*)(arow + k0 + 16 + 8 * half);
    v16bf a = __builtin_shufflevector(alo, ahi,
                0,1,2,3,4,5,6,7,8,9,10,11,12,13,14,15);
    #pragma unroll
    for (int ct = 0; ct < 4; ++ct) {
      v16bf bm = *(const v16bf*)(wpack + (((size_t)(k0 >> 5) * 4 + ct) * 32 + lane) * 16);
      acc[ct] = __builtin_amdgcn_wmma_f32_16x16x32_bf16(
          false, a, false, bm, (short)0, acc[ct], false, false);
    }
  }
  #pragma unroll
  for (int ct = 0; ct < 4; ++ct) {
    const int col = ct * 16 + lr;
    const float bv = bias[col];
    #pragma unroll
    for (int i = 0; i < 8; ++i) {             // C/D: VGPR i -> row half*8+i, col = lane
      const int row = row0 + half * 8 + i;
      hout[(size_t)row * 64 + col] = acc[ct][i] + bv;
    }
  }
}

// per-(node,head) attention logits for both relations
__global__ void k_att_node(const float* __restrict__ h,
                           const float* __restrict__ attSA, const float* __restrict__ attDA,
                           const float* __restrict__ attSP, const float* __restrict__ attDP,
                           float* __restrict__ asA, float* __restrict__ adA,
                           float* __restrict__ asP, float* __restrict__ adP,
                           int n) {
  int i = blockIdx.x * blockDim.x + threadIdx.x;
  int st = gridDim.x * blockDim.x;
  const int total = n * HEADS;
  for (; i < total; i += st) {
    const int hd = i & 7;
    const int node = i >> 3;
    const float* hp = h + (size_t)node * 64 + hd * 8;
    float v[8];
    #pragma unroll
    for (int d = 0; d < 8; ++d) v[d] = hp[d];
    float sSA = 0.f, sDA = 0.f, sSP = 0.f, sDP = 0.f;
    #pragma unroll
    for (int d = 0; d < 8; ++d) {
      sSA += v[d] * attSA[hd * 8 + d];
      sDA += v[d] * attDA[hd * 8 + d];
      sSP += v[d] * attSP[hd * 8 + d];
      sDP += v[d] * attDP[hd * 8 + d];
    }
    asA[i] = sSA; adA[i] = sDA; asP[i] = sSP; adP[i] = sDP;
  }
}

__global__ void k_edge_pass1(const int* __restrict__ src, const int* __restrict__ dst,
                             const float* __restrict__ asrc, const float* __restrict__ adst,
                             float* __restrict__ ealpha, unsigned* __restrict__ segmax, int E) {
  int i = blockIdx.x * blockDim.x + threadIdx.x;
  int st = gridDim.x * blockDim.x;
  const int total = E * 8;
  for (; i < total; i += st) {
    const int e = i >> 3, hd = i & 7;
    const int s = src[e], d = dst[e];
    float a = asrc[(size_t)s * 8 + hd] + adst[(size_t)d * 8 + hd];
    a = a > 0.f ? a : NEG_SLOPE * a;          // leaky_relu
    ealpha[i] = a;
    atomicMax(&segmax[(size_t)d * 8 + hd], enc_ordered(a));
  }
}

__global__ void k_edge_pass2(const int* __restrict__ dst,
                             float* __restrict__ ealpha,
                             const unsigned* __restrict__ segmax,
                             float* __restrict__ segsum, int E) {
  int i = blockIdx.x * blockDim.x + threadIdx.x;
  int st = gridDim.x * blockDim.x;
  const int total = E * 8;
  for (; i < total; i += st) {
    const int e = i >> 3, hd = i & 7;
    const int d = dst[e];
    const float m = dec_ordered(segmax[(size_t)d * 8 + hd]);
    const float ex = __expf(ealpha[i] - m);
    ealpha[i] = ex;
    atomicAdd(&segsum[(size_t)d * 8 + hd], ex);
  }
}

__global__ void k_edge_pass3(const int* __restrict__ src, const int* __restrict__ dst,
                             const float* __restrict__ ealpha,
                             const float* __restrict__ segsum,
                             const float* __restrict__ h,
                             float* __restrict__ out, int E) {
  int i = blockIdx.x * blockDim.x + threadIdx.x;
  int st = gridDim.x * blockDim.x;
  const int total = E * 8;
  for (; i < total; i += st) {
    const int e = i >> 3, hd = i & 7;
    const int s = src[e], d = dst[e];
    if (hd == 0) {                            // lookahead prefetch of a future src row
      const int e2 = e + 128;
      if (e2 < E) __builtin_prefetch(h + (size_t)src[e2] * 64, 0, 0);
    }
    const float w = ealpha[i] / (segsum[(size_t)d * 8 + hd] + 1e-16f);
    const float* hp = h + (size_t)s * 64 + hd * 8;
    float* op = out + (size_t)d * 64 + hd * 8;
    #pragma unroll
    for (int k = 0; k < 8; ++k) atomicAdd(op + k, hp[k] * w);
  }
}

// partial += sum_rows q . tanh(relu(o) @ kW + kb), kW pre-packed, WMMA GEMM.
// One wave computes a 16-row tile; wave32 shuffle reduction; 1 atomic per wave.
__global__ void k_score_wmma(const float* __restrict__ o,
                             const unsigned short* __restrict__ kwp,
                             const float* __restrict__ kb, const float* __restrict__ q,
                             float* __restrict__ scoreSlot, int n) {
  const int lane = threadIdx.x & 31;
  const int waveId = (blockIdx.x * blockDim.x + threadIdx.x) >> 5;
  const int waveStride = (gridDim.x * blockDim.x) >> 5;
  const int half = lane >> 4;
  const int lr   = lane & 15;
  const int ntiles = n >> 4;

  float part = 0.f;
  for (int tile = waveId; tile < ntiles; tile += waveStride) {
    const int row0 = tile << 4;
    const float* orow = o + (size_t)(row0 + lr) * 64;
    v8f acc[4] = {};
    #pragma unroll
    for (int ks = 0; ks < 2; ++ks) {
      const int k0 = ks * 32;
      v16bf a;
      #pragma unroll
      for (int j = 0; j < 8; ++j) {           // relu + bf16 convert in registers
        float v = orow[k0 + 8 * half + j];      v = v > 0.f ? v : 0.f;
        float w = orow[k0 + 16 + 8 * half + j]; w = w > 0.f ? w : 0.f;
        a[j]     = __builtin_bit_cast(__bf16, f32_to_bf16_bits(v));
        a[8 + j] = __builtin_bit_cast(__bf16, f32_to_bf16_bits(w));
      }
      #pragma unroll
      for (int ct = 0; ct < 4; ++ct) {
        v16bf bm = *(const v16bf*)(kwp + (((size_t)ks * 4 + ct) * 32 + lane) * 16);
        acc[ct] = __builtin_amdgcn_wmma_f32_16x16x32_bf16(
            false, a, false, bm, (short)0, acc[ct], false, false);
      }
    }
    #pragma unroll
    for (int ct = 0; ct < 4; ++ct) {
      const int col = ct * 16 + lr;
      const float kbv = kb[col], qv = q[col];
      float s = 0.f;
      #pragma unroll
      for (int i = 0; i < 8; ++i) s += tanhf(acc[ct][i] + kbv);
      part += qv * s;
    }
  }
  #pragma unroll
  for (int off = 16; off > 0; off >>= 1)      // wave32 reduction
    part += __shfl_xor(part, off, 32);
  if (lane == 0) atomicAdd(scoreSlot, part);
}

__global__ void k_beta(const float* __restrict__ scores, float* __restrict__ beta,
                       float invN) {
  const float s0 = scores[0] * invN, s1 = scores[1] * invN;  // mean over nodes
  const float m = fmaxf(s0, s1);
  const float e0 = __expf(s0 - m), e1 = __expf(s1 - m);
  const float inv = 1.f / (e0 + e1);
  beta[0] = e0 * inv; beta[1] = e1 * inv;
}

// y = LN(beta0*relu(out0) + beta1*relu(out1)); also emit bf16 copy for next WMMA
__global__ void k_blend_ln(const float* __restrict__ o0, const float* __restrict__ o1,
                           const float* __restrict__ beta,
                           const float* __restrict__ lnw, const float* __restrict__ lnb,
                           float* __restrict__ hout, unsigned short* __restrict__ hbf,
                           int n, int writeBf) {
  int row = blockIdx.x * blockDim.x + threadIdx.x;
  int st = gridDim.x * blockDim.x;
  const float b0 = beta[0], b1 = beta[1];
  for (; row < n; row += st) {
    const float* p0 = o0 + (size_t)row * 64;
    const float* p1 = o1 + (size_t)row * 64;
    float r[64]; float mu = 0.f;
    #pragma unroll
    for (int i = 0; i < 64; ++i) {
      float a = p0[i]; a = a > 0.f ? a : 0.f;
      float c = p1[i]; c = c > 0.f ? c : 0.f;
      const float v = b0 * a + b1 * c;
      r[i] = v; mu += v;
    }
    mu *= (1.0f / 64.0f);
    float var = 0.f;
    #pragma unroll
    for (int i = 0; i < 64; ++i) { const float d = r[i] - mu; var += d * d; }
    var *= (1.0f / 64.0f);
    const float inv = rsqrtf(var + LN_EPS);
    float* ho = hout + (size_t)row * 64;
    unsigned short* hb = hbf + (size_t)row * 64;
    #pragma unroll
    for (int i = 0; i < 64; ++i) {
      const float y = (r[i] - mu) * inv * lnw[i] + lnb[i];
      ho[i] = y;
      if (writeBf) hb[i] = f32_to_bf16_bits(y);
    }
  }
}

__global__ void k_final(const float* __restrict__ h, const float* __restrict__ W,
                        const float* __restrict__ b, float* __restrict__ out, int n) {
  int row = blockIdx.x * blockDim.x + threadIdx.x;
  int st = gridDim.x * blockDim.x;
  const float c0 = b[0], c1 = b[1], c2 = b[2];
  for (; row < n; row += st) {
    const float* hp = h + (size_t)row * 64;
    float l0 = c0, l1 = c1, l2 = c2;
    #pragma unroll
    for (int i = 0; i < 64; ++i) {
      const float v = hp[i];
      l0 += v * W[i * 3 + 0]; l1 += v * W[i * 3 + 1]; l2 += v * W[i * 3 + 2];
    }
    const float m = fmaxf(l0, fmaxf(l1, l2));
    const float e0 = __expf(l0 - m), e1 = __expf(l1 - m), e2 = __expf(l2 - m);
    const float inv = 1.f / (e0 + e1 + e2);
    out[(size_t)row * 3 + 0] = e0 * inv;
    out[(size_t)row * 3 + 1] = e1 * inv;
    out[(size_t)row * 3 + 2] = e2 * inv;
  }
}

// ------------------------------ host ------------------------------

extern "C" void kernel_launch(void* const* d_in, const int* in_sizes, int n_in,
                              void* d_out, int out_size, void* d_ws, size_t ws_size,
                              hipStream_t stream) {
  (void)n_in; (void)out_size; (void)ws_size;
  const float* x   = (const float*)d_in[0];
  const int*   eiA = (const int*)d_in[1];
  const int*   eiP = (const int*)d_in[2];
  const int n  = in_sizes[0] / 32;
  const int EA = in_sizes[1] / 2;
  const int EP = in_sizes[2] / 2;

  struct ConvP { const float *W,*b,*aSA,*aDA,*aSP,*aDP,*kW,*kb,*q; int K; };
  ConvP cv[3];
  const int Ks[3] = {32, 64, 64};
  for (int l = 0; l < 3; ++l) {
    const int base = 3 + l * 9;   // W, b, att_src_adj, att_dst_adj, att_src_pairs, att_dst_pairs, k_W, k_b, q
    cv[l].W   = (const float*)d_in[base + 0];
    cv[l].b   = (const float*)d_in[base + 1];
    cv[l].aSA = (const float*)d_in[base + 2];
    cv[l].aDA = (const float*)d_in[base + 3];
    cv[l].aSP = (const float*)d_in[base + 4];
    cv[l].aDP = (const float*)d_in[base + 5];
    cv[l].kW  = (const float*)d_in[base + 6];
    cv[l].kb  = (const float*)d_in[base + 7];
    cv[l].q   = (const float*)d_in[base + 8];
    cv[l].K   = Ks[l];
  }
  const float* lnw[3] = {(const float*)d_in[30], (const float*)d_in[32], (const float*)d_in[34]};
  const float* lnb[3] = {(const float*)d_in[31], (const float*)d_in[33], (const float*)d_in[35]};
  const float* linW = (const float*)d_in[36];
  const float* linb = (const float*)d_in[37];

  // ---- workspace carve (float units, 256-float aligned regions) ----
  float* wsf = (float*)d_ws;
  size_t off = 0;
  auto carve = [&](size_t cnt) {
    float* p = wsf + off;
    off += (cnt + 255) & ~(size_t)255;
    return p;
  };
  float* hf   = carve((size_t)n * 64);                    // projected h / next input / LN3 out
  float* out0 = carve((size_t)n * 64);                    // adjacent aggregation
  float* out1 = carve((size_t)n * 64);                    // pairs aggregation
  unsigned short* xbf = (unsigned short*)carve((size_t)n * 32);  // n*64 bf16 (WMMA A)
  float* asA = carve((size_t)n * 8);
  float* adA = carve((size_t)n * 8);
  float* asP = carve((size_t)n * 8);
  float* adP = carve((size_t)n * 8);
  unsigned* segmax = (unsigned*)carve((size_t)n * 8);
  float* segsum    = carve((size_t)n * 8);
  const int Emax = EA > EP ? EA : EP;
  float* ealpha = carve((size_t)Emax * 8);
  unsigned short* wpack  = (unsigned short*)carve(2048);  // packed proj weights (<=4096 bf16)
  unsigned short* kwpack = (unsigned short*)carve(2048);  // packed k_W (4096 bf16)
  float* scores = carve(64);
  float* beta   = scores + 8;

  const int TPB = 256;
  auto blks = [&](long long total) { return (int)((total + TPB - 1) / TPB); };

  // layer-1 input -> bf16
  k_f32_to_bf16<<<blks((long long)n * 32), TPB, 0, stream>>>(x, xbf, n * 32);

  for (int l = 0; l < 3; ++l) {
    const ConvP& c = cv[l];
    const int K = c.K;
    k_pack_b<<<blks(K * 64), TPB, 0, stream>>>(c.W, wpack, K);
    k_pack_b<<<blks(64 * 64), TPB, 0, stream>>>(c.kW, kwpack, 64);

    const int waves = (n + 15) / 16;
    k_proj_wmma<<<(waves + 7) / 8, TPB, 0, stream>>>(xbf, wpack, c.b, hf, n, K);

    k_att_node<<<blks((long long)n * 8), TPB, 0, stream>>>(
        hf, c.aSA, c.aDA, c.aSP, c.aDP, asA, adA, asP, adP, n);

    for (int et = 0; et < 2; ++et) {
      const int E = (et == 0) ? EA : EP;
      const int* src = (et == 0) ? eiA : eiP;
      const int* dst = src + E;
      float* outb = (et == 0) ? out0 : out1;
      const float* as = (et == 0) ? asA : asP;
      const float* ad = (et == 0) ? adA : adP;
      k_zero<<<blks((long long)n * 64), TPB, 0, stream>>>(outb, n * 64);
      k_zero<<<blks((long long)n * 8), TPB, 0, stream>>>((float*)segmax, n * 8);
      k_zero<<<blks((long long)n * 8), TPB, 0, stream>>>(segsum, n * 8);
      k_edge_pass1<<<blks((long long)E * 8), TPB, 0, stream>>>(src, dst, as, ad, ealpha, segmax, E);
      k_edge_pass2<<<blks((long long)E * 8), TPB, 0, stream>>>(dst, ealpha, segmax, segsum, E);
      k_edge_pass3<<<blks((long long)E * 8), TPB, 0, stream>>>(src, dst, ealpha, segsum, hf, outb, E);
    }

    k_zero<<<1, 64, 0, stream>>>(scores, 2);
    k_score_wmma<<<512, TPB, 0, stream>>>(out0, kwpack, c.kb, c.q, scores + 0, n);
    k_score_wmma<<<512, TPB, 0, stream>>>(out1, kwpack, c.kb, c.q, scores + 1, n);
    k_beta<<<1, 1, 0, stream>>>(scores, beta, 1.0f / (float)n);

    k_blend_ln<<<blks(n), TPB, 0, stream>>>(out0, out1, beta, lnw[l], lnb[l],
                                            hf, xbf, n, (l < 2) ? 1 : 0);
  }

  k_final<<<blks(n), TPB, 0, stream>>>(hf, linW, linb, (float*)d_out, n);
}
